// GCN_32066225832643
// MI455X (gfx1250) — matrix operand order repaired and checked by reading, not compiled, not run on hardware
//
#include <hip/hip_runtime.h>
#include <math.h>

// ---------------- Problem constants (from reference) ----------------
#define N0_SRC 400000
#define N0_DST 32000
#define E0_CNT 320000
#define N1_DST 3200
#define E1_CNT 32000
#define N2_DST 1024
#define E2_CNT 10240
#define FEAT   256        // IN_F == HID == 256
#define NCLS   47
#define NCLSP  48         // padded to multiple of 16

typedef __attribute__((ext_vector_type(16))) __bf16 v16bf;
typedef __attribute__((ext_vector_type(8)))  float  v8f;

// ---------------- Zero fill (grid-stride) ----------------
__global__ void gcn_fill0(float* __restrict__ p, size_t n) {
    size_t i = (size_t)blockIdx.x * blockDim.x + threadIdx.x;
    size_t stride = (size_t)gridDim.x * blockDim.x;
    for (; i < n; i += stride) p[i] = 0.0f;
}

// ---------------- Weight transpose + f32->bf16 ----------------
// WT[n*K + k] = (n < N) ? W[k*N + n] : 0     (W is row-major K x N)
__global__ void gcn_wt_bf16(const float* __restrict__ W, __bf16* __restrict__ WT,
                            int K, int N, int Npad) {
    int i = blockIdx.x * blockDim.x + threadIdx.x;
    if (i >= Npad * K) return;
    int n = i / K;
    int k = i - n * K;
    float v = (n < N) ? W[(size_t)k * N + n] : 0.0f;
    WT[i] = (__bf16)v;
}

__global__ void gcn_padbias(const float* __restrict__ b, float* __restrict__ bp,
                            int N, int Npad) {
    int i = threadIdx.x;
    if (i < Npad) bp[i] = (i < N) ? b[i] : 0.0f;
}

// ---------------- Edge scatter: one wave per edge ----------------
// Lane l handles features [8l, 8l+8): two coalesced float4 loads + 8 f32 atomics.
__global__ void gcn_scatter(const float* __restrict__ h, const int* __restrict__ src,
                            const int* __restrict__ dst, float* __restrict__ agg,
                            float* __restrict__ deg, int E) {
    int wid  = (blockIdx.x * blockDim.x + threadIdx.x) >> 5;
    int lane = threadIdx.x & 31;
    if (wid >= E) return;
    int s = src[wid];
    int d = dst[wid];
    const float4* ps = (const float4*)(h + (size_t)s * FEAT) + lane * 2;
    float4 v0 = ps[0];
    float4 v1 = ps[1];
    float* pd = agg + (size_t)d * FEAT + lane * 8;
    atomicAdd(pd + 0, v0.x); atomicAdd(pd + 1, v0.y);
    atomicAdd(pd + 2, v0.z); atomicAdd(pd + 3, v0.w);
    atomicAdd(pd + 4, v1.x); atomicAdd(pd + 5, v1.y);
    atomicAdd(pd + 6, v1.z); atomicAdd(pd + 7, v1.w);
    if (lane == 0) atomicAdd(deg + d, 1.0f);
}

// ---------------- Degree-normalize + convert to bf16 ----------------
__global__ void gcn_norm_bf16(const float* __restrict__ agg, const float* __restrict__ deg,
                              __bf16* __restrict__ out, int n /* rows*FEAT */) {
    int i = blockIdx.x * blockDim.x + threadIdx.x;
    if (i >= n) return;
    float s = 1.0f / fmaxf(deg[i >> 8], 1.0f);
    out[i] = (__bf16)(agg[i] * s);
}

// ---------------- WMMA GEMM: one wave = one 16x16 tile, K=256 ----------------
// A: [M x 256] bf16 row-major.  BT: [Npad x 256] bf16 (transposed weights).
// out[m*ldout + n] = relu?(sum_k A[m,k]*W[k,n] + bias[n])
// Operand layouts per CDNA5 ISA 7.12.2 (wave32):
//   A: lane(0-15)=M, khalf=lane>>4; elems e<8 -> k=c*32+khalf*8+e ; e>=8 -> +16
//   B: lane(0-15)=N, khalf=lane>>4; elems e<8 -> k=c*32+khalf*8+e ; e>=8 -> +16
//   C/D: col n = lane&15; VGPR r -> row m = 8*(lane>>4)+r
__global__ __launch_bounds__(256)
void gcn_gemm_wmma(const __bf16* __restrict__ A, const __bf16* __restrict__ BT,
                   const float* __restrict__ bias, float* __restrict__ out,
                   int tilesN, int ldout, int doRelu) {
    int wid  = (blockIdx.x * blockDim.x + threadIdx.x) >> 5;
    int lane = threadIdx.x & 31;
    int mt = wid / tilesN;
    int nt = wid - mt * tilesN;
    int half = lane >> 4;
    int l15  = lane & 15;

    const __bf16* arow = A  + (size_t)(mt * 16 + l15) * FEAT + half * 8;
    const __bf16* brow = BT + (size_t)(nt * 16 + l15) * FEAT + half * 8;

    v8f acc = {0.f, 0.f, 0.f, 0.f, 0.f, 0.f, 0.f, 0.f};
    union { v16bf v; uint4 q[2]; } a, b;

#pragma unroll
    for (int c = 0; c < 8; ++c) {            // K = 8 * 32 = 256
        a.q[0] = *(const uint4*)(arow + c * 32);
        a.q[1] = *(const uint4*)(arow + c * 32 + 16);
        b.q[0] = *(const uint4*)(brow + c * 32);
        b.q[1] = *(const uint4*)(brow + c * 32 + 16);
        acc = __builtin_amdgcn_wmma_f32_16x16x32_bf16(
            /*neg_a=*/false, a.v, /*neg_b=*/false, b.v,
            /*c_mod=*/(short)0, acc, /*reuse_a=*/false, /*reuse_b=*/false);
    }

    int ncol  = nt * 16 + l15;
    float bv  = bias[ncol];
    int mbase = mt * 16 + half * 8;
#pragma unroll
    for (int r = 0; r < 8; ++r) {
        float v = acc[r] + bv;
        if (doRelu) v = fmaxf(v, 0.0f);
        out[(size_t)(mbase + r) * ldout + ncol] = v;
    }
}

// ---------------- log_softmax over 47 classes: one wave per row ----------------
__global__ void gcn_logsoftmax(const float* __restrict__ logits, float* __restrict__ out) {
    int wid  = (blockIdx.x * blockDim.x + threadIdx.x) >> 5;
    int lane = threadIdx.x & 31;
    const float* row = logits + (size_t)wid * NCLSP;
    float a = row[lane];                                   // lane < 32 <= 47
    float b = (lane + 32 < NCLS) ? row[lane + 32] : -3.402823466e38f;
    float m = fmaxf(a, b);
#pragma unroll
    for (int o = 16; o >= 1; o >>= 1) m = fmaxf(m, __shfl_xor(m, o, 32));
    float e = expf(a - m) + ((lane + 32 < NCLS) ? expf(b - m) : 0.0f);
#pragma unroll
    for (int o = 16; o >= 1; o >>= 1) e += __shfl_xor(e, o, 32);
    float l = m + logf(e);
    out[(size_t)wid * NCLS + lane] = a - l;
    if (lane + 32 < NCLS) out[(size_t)wid * NCLS + lane + 32] = b - l;
}

// ---------------- Host-side orchestration ----------------
extern "C" void kernel_launch(void* const* d_in, const int* in_sizes, int n_in,
                              void* d_out, int out_size, void* d_ws, size_t ws_size,
                              hipStream_t stream) {
    const float* x    = (const float*)d_in[0];
    const float* W0   = (const float*)d_in[1];
    const float* b0   = (const float*)d_in[2];
    const float* W1   = (const float*)d_in[3];
    const float* b1   = (const float*)d_in[4];
    const float* W2   = (const float*)d_in[5];
    const float* b2   = (const float*)d_in[6];
    const int*   src0 = (const int*)d_in[7];
    const int*   dst0 = (const int*)d_in[8];
    const int*   src1 = (const int*)d_in[9];
    const int*   dst1 = (const int*)d_in[10];
    const int*   src2 = (const int*)d_in[11];
    const int*   dst2 = (const int*)d_in[12];

    char* ws = (char*)d_ws;
    size_t off = 0;
    auto alloc = [&](size_t bytes) -> char* {
        char* p = ws + off;
        off = (off + bytes + 255) & ~(size_t)255;
        return p;
    };

    // Buffers that must be zeroed every call (contiguous region first).
    float* agg0 = (float*)alloc((size_t)N0_DST * FEAT * 4);
    float* deg0 = (float*)alloc((size_t)N0_DST * 4);
    float* agg1 = (float*)alloc((size_t)N1_DST * FEAT * 4);
    float* deg1 = (float*)alloc((size_t)N1_DST * 4);
    float* agg2 = (float*)alloc((size_t)N2_DST * FEAT * 4);
    float* deg2 = (float*)alloc((size_t)N2_DST * 4);
    size_t zero_floats = off / 4;

    __bf16* Xb0  = (__bf16*)alloc((size_t)N0_DST * FEAT * 2);
    __bf16* Xb1  = (__bf16*)alloc((size_t)N1_DST * FEAT * 2);
    __bf16* Xb2  = (__bf16*)alloc((size_t)N2_DST * FEAT * 2);
    __bf16* WT0  = (__bf16*)alloc((size_t)FEAT * FEAT * 2);
    __bf16* WT1  = (__bf16*)alloc((size_t)FEAT * FEAT * 2);
    __bf16* WT2  = (__bf16*)alloc((size_t)NCLSP * FEAT * 2);
    float*  b2p  = (float*)alloc((size_t)NCLSP * 4);
    float*  logits = (float*)alloc((size_t)N2_DST * NCLSP * 4);

    // H0 aliases agg0, H1 aliases agg1 (agg buffers are dead after normalize).
    float* H0 = agg0;
    float* H1 = agg1;

    const int B = 256;

    // 1) Zero accumulators + degrees.
    gcn_fill0<<<2048, B, 0, stream>>>((float*)ws, zero_floats);

    // 2) Weight conversion (transposed bf16) + padded bias.
    gcn_wt_bf16<<<(FEAT * FEAT + B - 1) / B, B, 0, stream>>>(W0, WT0, FEAT, FEAT, FEAT);
    gcn_wt_bf16<<<(FEAT * FEAT + B - 1) / B, B, 0, stream>>>(W1, WT1, FEAT, FEAT, FEAT);
    gcn_wt_bf16<<<(NCLSP * FEAT + B - 1) / B, B, 0, stream>>>(W2, WT2, FEAT, NCLS, NCLSP);
    gcn_padbias<<<1, 64, 0, stream>>>(b2, b2p, NCLS, NCLSP);

    // ---- Layer 0 ----
    gcn_scatter<<<(E0_CNT * 32 + B - 1) / B, B, 0, stream>>>(x, src0, dst0, agg0, deg0, E0_CNT);
    gcn_norm_bf16<<<(N0_DST * FEAT + B - 1) / B, B, 0, stream>>>(agg0, deg0, Xb0, N0_DST * FEAT);
    gcn_gemm_wmma<<<(N0_DST / 16) * (FEAT / 16) * 32 / B, B, 0, stream>>>(
        Xb0, WT0, b0, H0, FEAT / 16, FEAT, 1);

    // ---- Layer 1 ----
    gcn_scatter<<<(E1_CNT * 32 + B - 1) / B, B, 0, stream>>>(H0, src1, dst1, agg1, deg1, E1_CNT);
    gcn_norm_bf16<<<(N1_DST * FEAT + B - 1) / B, B, 0, stream>>>(agg1, deg1, Xb1, N1_DST * FEAT);
    gcn_gemm_wmma<<<(N1_DST / 16) * (FEAT / 16) * 32 / B, B, 0, stream>>>(
        Xb1, WT1, b1, H1, FEAT / 16, FEAT, 1);

    // ---- Layer 2 (no ReLU) ----
    gcn_scatter<<<(E2_CNT * 32 + B - 1) / B, B, 0, stream>>>(H1, src2, dst2, agg2, deg2, E2_CNT);
    gcn_norm_bf16<<<(N2_DST * FEAT + B - 1) / B, B, 0, stream>>>(agg2, deg2, Xb2, N2_DST * FEAT);
    gcn_gemm_wmma<<<(N2_DST / 16) * (NCLSP / 16) * 32 / B, B, 0, stream>>>(
        Xb2, WT2, b2p, logits, NCLSP / 16, NCLSP, 0);

    // ---- log_softmax -> d_out [1024 x 47] f32 ----
    gcn_logsoftmax<<<N2_DST * 32 / B, B, 0, stream>>>(logits, (float*)d_out);
}
